// Head_22222160790117
// MI455X (gfx1250) — matrix operand order repaired and checked by reading, not compiled, not run on hardware
//
#include <hip/hip_runtime.h>

// ---------------- CDNA5 / gfx1250 wave32 WMMA attention ----------------
// q = x@Wq, k = x@Wk stored bf16 row-major [b*T + t][H]
// vT = (x@Wv)^T stored bf16 [b*H + h][T]   (computed directly as Wv^T @ x^T)
// P  = softmax(mask(q k^T / sqrt(32)))  bf16 [b*T + t][T]
// out = P @ v  (via vT) fp32 [b*T + t][H]

constexpr int Bc = 8;
constexpr int Tc = 2048;
constexpr int Cc = 32;
constexpr int Hc = 2048;

typedef __attribute__((ext_vector_type(16))) __bf16 v16bf;
typedef __attribute__((ext_vector_type(8)))  float  v8f;

#define SC_LD 2052  // padded LDS row stride (floats) for score rows

static __device__ inline unsigned short f2bf(float f) {
  unsigned int u = __float_as_uint(f);
  u += 0x7FFFu + ((u >> 16) & 1u);  // round-to-nearest-even
  return (unsigned short)(u >> 16);
}

// Async global->LDS copy of one 16-byte chunk per lane (CDNA5 ASYNCcnt path).
static __device__ inline void async_copy_b128(unsigned ldsByteOff, const void* gaddr) {
  asm volatile("global_load_async_to_lds_b128 %0, %1, off"
               :: "v"(ldsByteOff), "v"((unsigned long long)(size_t)gaddr)
               : "memory");
}
static __device__ inline void async_wait0() {
  asm volatile("s_wait_asynccnt 0x0" ::: "memory");
}

// ---- fragment loaders (V_WMMA_F32_16X16X32_BF16 operands in this file's contiguous-K order, wave32) ----
// A 16x32 bf16: lane = M | half; elems 0..15 -> K = half*16 + 0..15  (per-lane contiguous K)
// B 32x16 bf16: lane = N | half; elems 0..15 -> K = half*16 + 0..15  (per-lane contiguous K)

// A fragment from an LDS-resident 16-row tile (row stride ld elements) -> ds_load_b128
static __device__ inline v16bf load_a_lds(const unsigned short* tile, int ld, int k0) {
  int lane = threadIdx.x & 31;
  int m = lane & 15, half = lane >> 4;
  const unsigned short* p = tile + m * ld + k0;
  union { v16bf v; uint4 q[2]; } f;
  f.q[0] = *(const uint4*)(p + half * 16);
  f.q[1] = *(const uint4*)(p + half * 16 + 8);
  return f.v;
}

static __device__ inline v16bf load_bT_bf16(const unsigned short* srcT, int ld, int n0, int k0) {
  int lane = threadIdx.x & 31;
  int n = lane & 15, kb = (lane >> 4) << 4;
  const unsigned short* p = srcT + (size_t)(n0 + n) * ld + k0 + kb;
  union { v16bf v; uint4 q[2]; } f;
  f.q[0] = *(const uint4*)(p);
  f.q[1] = *(const uint4*)(p + 8);
  return f.v;
}

static __device__ inline v16bf load_a_f32(const float* src, int ld, int m0) {
  int lane = threadIdx.x & 31;
  int m = lane & 15, half = lane >> 4;
  const float* p = src + (size_t)(m0 + m) * ld;
  union { v16bf v; unsigned short u[16]; } f;
#pragma unroll
  for (int e = 0; e < 8; ++e) f.u[e] = f2bf(p[half * 16 + e]);
#pragma unroll
  for (int e = 0; e < 8; ++e) f.u[8 + e] = f2bf(p[half * 16 + 8 + e]);
  return f.v;
}

// A fragment from a transposed fp32 source: A[m][k] = src[k][m0+m] (src row-major [K][ld])
static __device__ inline v16bf load_aT_f32(const float* src, int ld, int m0) {
  int lane = threadIdx.x & 31;
  int m = lane & 15, half = lane >> 4;
  const float* p = src + (m0 + m);
  union { v16bf v; unsigned short u[16]; } f;
#pragma unroll
  for (int e = 0; e < 8; ++e) f.u[e] = f2bf(p[(size_t)(half * 16 + e) * ld]);
#pragma unroll
  for (int e = 0; e < 8; ++e) f.u[8 + e] = f2bf(p[(size_t)(half * 16 + 8 + e) * ld]);
  return f.v;
}

// B fragment from fp32 row-major [K][ld] (column slice, strided)
static __device__ inline v16bf load_b_f32_cols(const float* src, int ld, int k0, int n0) {
  int lane = threadIdx.x & 31;
  int n = lane & 15, kb = (lane >> 4) << 4;
  const float* p = src + (size_t)(k0 + kb) * ld + (n0 + n);
  union { v16bf v; unsigned short u[16]; } f;
#pragma unroll
  for (int e = 0; e < 16; ++e) f.u[e] = f2bf(p[(size_t)e * ld]);
  return f.v;
}

// B fragment from fp32 where B[k][n] = src[n0+n][k] (per-lane contiguous K)
static __device__ inline v16bf load_b_f32_rows(const float* src, int ld, int n0, int k0) {
  int lane = threadIdx.x & 31;
  int n = lane & 15, kb = (lane >> 4) << 4;
  const float* p = src + (size_t)(n0 + n) * ld + k0 + kb;
  union { v16bf v; unsigned short u[16]; } f;
#pragma unroll
  for (int e = 0; e < 16; ++e) f.u[e] = f2bf(p[e]);
  return f.v;
}

// C/D 16x16 f32: VGPR r, lanes 0-15 -> (M=r, N=lane), lanes 16-31 -> (M=8+r, N=lane-16)
static __device__ inline void store_tile_bf16(unsigned short* dst, int ld, int m0, int n0, v8f acc) {
  int lane = threadIdx.x & 31;
  int n = lane & 15, half = lane >> 4;
#pragma unroll
  for (int r = 0; r < 8; ++r)
    dst[(size_t)(m0 + r + 8 * half) * ld + n0 + n] = f2bf(acc[r]);
}

static __device__ inline void store_tile_f32_nt(float* dst, int ld, int m0, int n0, v8f acc) {
  int lane = threadIdx.x & 31;
  int n = lane & 15, half = lane >> 4;
#pragma unroll
  for (int r = 0; r < 8; ++r)
    __builtin_nontemporal_store(acc[r], &dst[(size_t)(m0 + r + 8 * half) * ld + n0 + n]);
}

// ---------------- kernel 1: q,k projections (M=B*T, N=H, K=32) ----------------
__global__ void proj_qk(const float* __restrict__ x, const float* __restrict__ Wq,
                        const float* __restrict__ Wk,
                        unsigned short* __restrict__ qb, unsigned short* __restrict__ kb) {
  const int wid = threadIdx.x >> 5;
  const int m0 = blockIdx.x * 16;                 // flat b*T + t row
  const int n0 = blockIdx.y * 128 + wid * 16;     // h
  v16bf a  = load_a_f32(x, Cc, m0);
  v16bf bq = load_b_f32_cols(Wq, Hc, 0, n0);
  v16bf bk = load_b_f32_cols(Wk, Hc, 0, n0);
  v8f cq = {}, ck = {};
  cq = __builtin_amdgcn_wmma_f32_16x16x32_bf16(false, a, false, bq, (short)0, cq, false, false);
  ck = __builtin_amdgcn_wmma_f32_16x16x32_bf16(false, a, false, bk, (short)0, ck, false, false);
  store_tile_bf16(qb, Hc, m0, n0, cq);
  store_tile_bf16(kb, Hc, m0, n0, ck);
}

// ---------------- kernel 2: vT = Wv^T @ x^T   (M=H, N=T, K=32, per batch) ----------------
__global__ void proj_vT(const float* __restrict__ x, const float* __restrict__ Wv,
                        unsigned short* __restrict__ vtb) {
  const int wid = threadIdx.x >> 5;
  const int b  = blockIdx.z;
  const int m0 = blockIdx.x * 16;                 // h
  const int n0 = blockIdx.y * 128 + wid * 16;     // t
  v16bf a  = load_aT_f32(Wv, Hc, m0);             // A[h][c] = Wv[c][h]
  v16bf bx = load_b_f32_rows(x, Cc, b * Tc + n0, 0); // B[c][t] = x[t][c]
  v8f c = {};
  c = __builtin_amdgcn_wmma_f32_16x16x32_bf16(false, a, false, bx, (short)0, c, false, false);
  store_tile_bf16(vtb, Tc, b * Hc + m0, n0, c);
}

// ------- kernel 3: causal scores (WMMA over H) + row softmax in LDS -> P bf16 -------
// LDS: [16][Hc] bf16 staged q tile (async copy) + [16][SC_LD] fp32 score rows
__global__ void attn_scores_softmax(const unsigned short* __restrict__ qb,
                                    const unsigned short* __restrict__ kb,
                                    unsigned short* __restrict__ Pb) {
  extern __shared__ unsigned char smem[];
  unsigned short* qs = (unsigned short*)smem;                       // 16*Hc bf16 = 64KB
  float* sc = (float*)(smem + (size_t)16 * Hc * sizeof(unsigned short));

  const int tt = blockIdx.x, b = blockIdx.y;
  const int wid = threadIdx.x >> 5, lane = threadIdx.x & 31;
  const int t0 = tt * 16;
  const int rowBase = b * Tc;
  const float scale = 0.17677669529663687f;       // 1/sqrt(32)

  // --- async stage of the q tile: 16 consecutive rows = flat 64KB region ---
  {
    const unsigned short* gq = qb + (size_t)(rowBase + t0) * Hc;
    const unsigned ldsBase = (unsigned)(size_t)qs;   // LDS aperture: low 32 bits = LDS offset
    for (int c = threadIdx.x; c < (16 * Hc) / 8; c += 512) {
      async_copy_b128(ldsBase + c * 16, gq + c * 8);
    }
    async_wait0();
  }
  __syncthreads();

  // 16 waves split the s-tiles (causal: only st <= tt)
  for (int st = wid; st <= tt; st += 16) {
    v8f acc = {};
    for (int hk = 0; hk < Hc; hk += 32) {
      __builtin_prefetch(kb + (size_t)(rowBase + st * 16) * Hc + hk + 64, 0, 1);
      v16bf a   = load_a_lds(qs, Hc, hk);
      v16bf bfr = load_bT_bf16(kb, Hc, rowBase + st * 16, hk);
      acc = __builtin_amdgcn_wmma_f32_16x16x32_bf16(false, a, false, bfr, (short)0, acc, false, false);
    }
    int n = lane & 15, half = lane >> 4;
#pragma unroll
    for (int r = 0; r < 8; ++r) {
      int M = r + 8 * half;
      float v = acc[r] * scale;
      if (st == tt && n > M) v = -__builtin_inff();   // causal mask on diagonal tile
      sc[M * SC_LD + st * 16 + n] = v;
    }
  }
  __syncthreads();

  // softmax: wave `wid` owns query row t0+wid
  {
    const int m = wid;
    const int t = t0 + m;
    const int ncols = (tt + 1) * 16;
    float* row = sc + m * SC_LD;
    float mx = -__builtin_inff();
    for (int s = lane; s < ncols; s += 32) mx = fmaxf(mx, row[s]);
#pragma unroll
    for (int off = 16; off; off >>= 1) mx = fmaxf(mx, __shfl_xor(mx, off));
    float sum = 0.f;
    for (int s = lane; s < ncols; s += 32) { float e = __expf(row[s] - mx); sum += e; row[s] = e; }
#pragma unroll
    for (int off = 16; off; off >>= 1) sum += __shfl_xor(sum, off);
    float inv = 1.0f / sum;
    unsigned short* prow = Pb + (size_t)(b * Tc + t) * Tc;
    for (int s = lane; s < ncols; s += 32) prow[s] = f2bf(row[s] * inv);
    // zero-pad P up to the next 32-column boundary for the K=32 PV loop
    int pad = (ncols + 31) & ~31;
    for (int s = ncols + lane; s < pad; s += 32) prow[s] = 0;
  }
}

// ---------------- kernel 4: out = P @ V  via vT (M=T, N=H, K<=T causal) ----------------
// LDS: [16][Tc] bf16 staged P tile (async copy), shared by the block's 8 waves.
__global__ void attn_pv(const unsigned short* __restrict__ Pb,
                        const unsigned short* __restrict__ vtb,
                        float* __restrict__ out) {
  extern __shared__ unsigned char smem[];
  unsigned short* ps = (unsigned short*)smem;      // 16*Tc bf16 = 64KB (only `pad` cols used)

  const int wid = threadIdx.x >> 5;
  const int tt = blockIdx.x, b = blockIdx.z;
  const int t0 = tt * 16;
  const int h0 = blockIdx.y * 128 + wid * 16;
  const int pad = ((tt + 1) * 16 + 31) & ~31;     // causal K extent, rounded to 32

  // --- async stage of the P tile: 16 rows x pad bf16 cols ---
  {
    const unsigned ldsBase = (unsigned)(size_t)ps;
    const int cpr = pad >> 3;                     // 16-byte chunks per row
    for (int c = threadIdx.x; c < 16 * cpr; c += 256) {
      int r = c / cpr, col = (c - r * cpr) * 8;
      async_copy_b128(ldsBase + (unsigned)(r * Tc + col) * 2,
                      Pb + (size_t)(b * Tc + t0 + r) * Tc + col);
    }
    async_wait0();
  }
  __syncthreads();

  v8f acc = {};
  for (int sk = 0; sk < pad; sk += 32) {
    v16bf a   = load_a_lds(ps, Tc, sk);
    v16bf bfr = load_bT_bf16(vtb, Tc, b * Hc + h0, sk);
    acc = __builtin_amdgcn_wmma_f32_16x16x32_bf16(false, a, false, bfr, (short)0, acc, false, false);
  }
  store_tile_f32_nt(out, Hc, b * Tc + t0, h0, acc);
}

// ---------------- host launch ----------------
extern "C" void kernel_launch(void* const* d_in, const int* in_sizes, int n_in,
                              void* d_out, int out_size, void* d_ws, size_t ws_size,
                              hipStream_t stream) {
  const float* x  = (const float*)d_in[0];
  const float* Wk = (const float*)d_in[1];
  const float* Wq = (const float*)d_in[2];
  const float* Wv = (const float*)d_in[3];
  float* out = (float*)d_out;

  const size_t qE = (size_t)Bc * Tc * Hc;         // 33.5M elems
  unsigned short* qb   = (unsigned short*)d_ws;   // bf16 q   [B*T][H]
  unsigned short* kbuf = qb + qE;                 // bf16 k   [B*T][H]
  unsigned short* vtb  = kbuf + qE;               // bf16 v^T [B*H][T]
  unsigned short* Pb   = vtb + qE;                // bf16 P   [B*T][T]

  const int smemScores = 16 * Hc * 2 + 16 * SC_LD * 4;   // 64KB q + 128.3KB scores
  const int smemPV     = 16 * Tc * 2;                     // 64KB P tile

  proj_qk<<<dim3(Bc * Tc / 16, Hc / 128), 256, 0, stream>>>(x, Wq, Wk, qb, kbuf);
  proj_vT<<<dim3(Hc / 16, Tc / 128, Bc), 256, 0, stream>>>(x, Wv, vtb);
  attn_scores_softmax<<<dim3(Tc / 16, Bc), 512, smemScores, stream>>>(qb, kbuf, Pb);
  attn_pv<<<dim3(Tc / 16, Hc / 128, Bc), 256, smemPV, stream>>>(Pb, vtb, out);
}
